// GIN_41154376630503
// MI455X (gfx1250) — matrix operand order repaired and checked by reading, not compiled, not run on hardware
//
#include <hip/hip_runtime.h>

#define D   128
#define STR 136   // padded LDS row stride (bf16 elements): 272B = 68 banks -> conflict-free
#define BM  128   // rows of z per block

typedef __attribute__((ext_vector_type(16))) __bf16         v16bf;
typedef __attribute__((ext_vector_type(8)))  float          v8f;
typedef __attribute__((ext_vector_type(8)))  unsigned short v8u;

struct __align__(32) BFrag { v8u lo, hi; };

static __device__ __forceinline__ v16bf as_bf16(BFrag f) {
  return __builtin_bit_cast(v16bf, f);
}

static __device__ __forceinline__ unsigned short f2bf(float f) {
  unsigned int u = __builtin_bit_cast(unsigned int, f);
  u += 0x7FFFu + ((u >> 16) & 1u);           // round-to-nearest-even
  return (unsigned short)(u >> 16);
}

// ---------------------------------------------------------------- zero fill
__global__ void zero_f32(float* __restrict__ p, long n) {
  long i = (long)blockIdx.x * blockDim.x + threadIdx.x;
  if (i < n) p[i] = 0.0f;
}

// ------------------------------------------------- edge scatter-add (L2-resident)
__global__ void gin_scatter(const float* __restrict__ h,
                            const int* __restrict__ src,
                            const int* __restrict__ dst,
                            float* __restrict__ agg, int E_) {
  int idx  = blockIdx.x * blockDim.x + threadIdx.x;
  int e    = idx >> 5;
  if (e >= E_) return;
  int c    = (idx & 31) * 4;                  // 32 lanes x float4 = 128 feats
  int s    = src[e], d = dst[e];
  const float4 v = *(const float4*)(h + (long)s * D + c);
  float* o = agg + (long)d * D + c;
  atomicAdd(o + 0, v.x); atomicAdd(o + 1, v.y);
  atomicAdd(o + 2, v.z); atomicAdd(o + 3, v.w);
}

// ------------- fused: z=(1+eps)h+agg -> relu(zW1+b1) -> relu(.W2+b2) -> BN ------------
__global__ __launch_bounds__(256) void gin_mlp(
    const float* __restrict__ h,   const float* __restrict__ agg,
    const float* __restrict__ W1,  const float* __restrict__ b1,
    const float* __restrict__ W2,  const float* __restrict__ b2,
    const float* __restrict__ gam, const float* __restrict__ bet,
    const float* __restrict__ mean,const float* __restrict__ var,
    const float* __restrict__ eps, int layer,
    float* __restrict__ hout, int nrows)
{
  extern __shared__ unsigned short smem[];
  unsigned short* zin  = smem;                     // BM*STR  (bf16 z)
  unsigned short* zmid = smem + 1 * BM * STR;      // BM*STR  (bf16 relu(zW1+b1))
  unsigned short* w1t  = smem + 2 * BM * STR;      // D*STR   W1^T [n][k]
  unsigned short* w2t  = smem + 3 * BM * STR;      // D*STR   W2^T [n][k]

  const int tid     = threadIdx.x;
  const int m0block = blockIdx.x * BM;
  const float epl   = 1.0f + eps[layer];

  // cooperative: load + transpose + bf16-convert both weight matrices
  for (int i = tid; i < D * D; i += 256) {
    int k = i >> 7, n = i & 127;                  // W[k][n] row-major, coalesced read
    w1t[n * STR + k] = f2bf(W1[i]);
    w2t[n * STR + k] = f2bf(W2[i]);
  }
  // cooperative: z = (1+eps)*h + agg  (rows past end -> 0)
  for (int i = tid; i < BM * D; i += 256) {
    int r = i >> 7, c = i & 127;
    int g = m0block + r;
    float v = 0.0f;
    if (g < nrows) v = epl * h[(long)g * D + c] + agg[(long)g * D + c];
    zin[r * STR + c] = f2bf(v);
  }
  __syncthreads();

  const int lane = tid & 31;
  const int m0   = (tid >> 5) * 16;               // wave's 16-row stripe
  const int lrow = lane & 15;
  const int kh   = lane >> 4;

  // ---- GEMM1: preload 4 A fragments (16x32 bf16 layout, ISA 7.12.2) ----
  BFrag a[4];
  {
    const unsigned short* base = zin + (m0 + lrow) * STR;
#pragma unroll
    for (int kt = 0; kt < 4; ++kt) {
      a[kt].lo = *(const v8u*)(base + kt * 32 + kh * 8);        // K = k0+8kh..+7
      a[kt].hi = *(const v8u*)(base + kt * 32 + 16 + kh * 8);   // K = k0+16+8kh..+7
    }
  }
#pragma unroll
  for (int nt = 0; nt < 8; ++nt) {
    v8f c = {};
    const int n = nt * 16 + lrow;
    const unsigned short* wb = w1t + n * STR + kh * 8;          // B: K = k0+8kh+j, k0+16+8kh+j
#pragma unroll
    for (int kt = 0; kt < 4; ++kt) {
      BFrag b;
      b.lo = *(const v8u*)(wb + kt * 32);
      b.hi = *(const v8u*)(wb + kt * 32 + 16);
      c = __builtin_amdgcn_wmma_f32_16x16x32_bf16(
              false, as_bf16(a[kt]), false, as_bf16(b), (short)0, c, false, false);
    }
    const float bias = b1[n];
#pragma unroll
    for (int r = 0; r < 8; ++r) {                               // D: m = r + 8*kh
      int m = m0 + r + kh * 8;
      float v = c[r] + bias;
      zmid[m * STR + n] = f2bf(v > 0.0f ? v : 0.0f);
    }
  }

  // ---- GEMM2: A comes from rows this same wave just wrote (no barrier needed) ----
  {
    const unsigned short* base = zmid + (m0 + lrow) * STR;
#pragma unroll
    for (int kt = 0; kt < 4; ++kt) {
      a[kt].lo = *(const v8u*)(base + kt * 32 + kh * 8);
      a[kt].hi = *(const v8u*)(base + kt * 32 + 16 + kh * 8);
    }
  }
#pragma unroll
  for (int nt = 0; nt < 8; ++nt) {
    v8f c = {};
    const int n = nt * 16 + lrow;
    const unsigned short* wb = w2t + n * STR + kh * 8;
#pragma unroll
    for (int kt = 0; kt < 4; ++kt) {
      BFrag b;
      b.lo = *(const v8u*)(wb + kt * 32);
      b.hi = *(const v8u*)(wb + kt * 32 + 16);
      c = __builtin_amdgcn_wmma_f32_16x16x32_bf16(
              false, as_bf16(a[kt]), false, as_bf16(b), (short)0, c, false, false);
    }
    const float bias = b2[n];
    const float sc = gam[n] * rsqrtf(var[n] + 1e-5f);           // eval-mode BN
    const float sh = bet[n] - mean[n] * sc;
#pragma unroll
    for (int r = 0; r < 8; ++r) {
      int m = m0 + r + kh * 8;
      int g = m0block + m;
      float v = c[r] + bias;
      v = (v > 0.0f ? v : 0.0f) * sc + sh;
      if (g < nrows) hout[(long)g * D + n] = v;
    }
  }
}

// ---------------------------------------------------------------- pooling
__global__ void pool_count(const int* __restrict__ batch, float* __restrict__ counts, int n) {
  int i = blockIdx.x * blockDim.x + threadIdx.x;
  if (i < n) atomicAdd(&counts[batch[i]], 1.0f);
}

__global__ void pool_sum(const float* __restrict__ h, const int* __restrict__ batch,
                         float* __restrict__ pooled, int n) {
  int idx  = blockIdx.x * blockDim.x + threadIdx.x;
  int node = idx >> 5;
  if (node >= n) return;
  int c = (idx & 31) * 4;
  int g = batch[node];
  const float4 v = *(const float4*)(h + (long)node * D + c);
  float* o = pooled + (long)g * D + c;
  atomicAdd(o + 0, v.x); atomicAdd(o + 1, v.y);
  atomicAdd(o + 2, v.z); atomicAdd(o + 3, v.w);
}

__global__ void final_lin(const float* __restrict__ pooled, const float* __restrict__ counts,
                          const float* __restrict__ W, const float* __restrict__ b,
                          float* __restrict__ out) {
  int idx = blockIdx.x * blockDim.x + threadIdx.x;   // G*D threads
  int g = idx >> 7, n = idx & 127;
  float inv = 1.0f / fmaxf(counts[g], 1.0f);
  float acc = b[n];
  for (int k = 0; k < D; ++k)
    acc += (pooled[g * D + k] * inv) * W[k * D + n];
  out[idx] = fmaxf(acc, 0.0f);
}

// ---------------------------------------------------------------- driver
extern "C" void kernel_launch(void* const* d_in, const int* in_sizes, int n_in,
                              void* d_out, int out_size, void* d_ws, size_t ws_size,
                              hipStream_t stream) {
  const float* x    = (const float*)d_in[0];
  const int*   ei   = (const int*)  d_in[1];
  const int*   batch= (const int*)  d_in[2];
  const float* W1   = (const float*)d_in[3];
  const float* b1   = (const float*)d_in[4];
  const float* W2   = (const float*)d_in[5];
  const float* b2   = (const float*)d_in[6];
  const float* gam  = (const float*)d_in[7];
  const float* bet  = (const float*)d_in[8];
  const float* rm   = (const float*)d_in[9];
  const float* rv   = (const float*)d_in[10];
  const float* eps  = (const float*)d_in[11];
  const float* lW   = (const float*)d_in[12];
  const float* lb   = (const float*)d_in[13];

  const int Nn = in_sizes[0] / D;     // 100000 nodes
  const int Ee = in_sizes[1] / 2;     // 1600000 edges
  const int Gg = out_size / D;        // 512 graphs

  float* hA     = (float*)d_ws;
  float* hB     = hA  + (size_t)Nn * D;
  float* agg    = hB  + (size_t)Nn * D;
  float* pooled = agg + (size_t)Nn * D;
  float* counts = pooled + (size_t)Gg * D;     // contiguous after pooled

  const int* src = ei;
  const int* dst = ei + Ee;

  const size_t lds_bytes  = 4 * (size_t)BM * STR * sizeof(unsigned short); // ~136 KB / 320 KB WGP
  const int    mlp_blocks = (Nn + BM - 1) / BM;
  const long   nz         = (long)Nn * D;
  const long   scat_thr   = (long)Ee * 32;

  const float* hcur = x;
  float* bufs[2] = { hA, hB };
  for (int l = 0; l < 3; ++l) {
    zero_f32  <<<(int)((nz + 255) / 256), 256, 0, stream>>>(agg, nz);
    gin_scatter<<<(int)((scat_thr + 255) / 256), 256, 0, stream>>>(hcur, src, dst, agg, Ee);
    float* hnext = bufs[l & 1];
    gin_mlp<<<mlp_blocks, 256, lds_bytes, stream>>>(
        hcur, agg,
        W1 + (size_t)l * D * D, b1 + (size_t)l * D,
        W2 + (size_t)l * D * D, b2 + (size_t)l * D,
        gam + (size_t)l * D, bet + (size_t)l * D,
        rm  + (size_t)l * D, rv  + (size_t)l * D,
        eps, l, hnext, Nn);
    hcur = hnext;
  }

  long pz = (long)Gg * D + Gg;        // pooled + counts
  zero_f32  <<<(int)((pz + 255) / 256), 256, 0, stream>>>(pooled, pz);
  pool_count<<<(Nn + 255) / 256, 256, 0, stream>>>(batch, counts, Nn);
  pool_sum  <<<(int)(((long)Nn * 32 + 255) / 256), 256, 0, stream>>>(hcur, batch, pooled, Nn);
  final_lin <<<(Gg * D + 255) / 256, 256, 0, stream>>>(pooled, counts, lW, lb, (float*)d_out);
}
